// OnceAggregation_32899449487473
// MI455X (gfx1250) — matrix operand order repaired and checked by reading, not compiled, not run on hardware
//
#include <hip/hip_runtime.h>

typedef __attribute__((ext_vector_type(16))) __bf16 v16bf;
typedef __attribute__((ext_vector_type(8)))  __bf16 v8bf;
typedef __attribute__((ext_vector_type(8)))  float  v8f;

#define WAVES 2   // waves per block; each wave owns a 32-row (2 x 16) tile

// ---------- helpers ----------

__device__ __forceinline__ unsigned short f2bf(float f) {
  unsigned int x = __float_as_uint(f);
  x = x + 0x7FFFu + ((x >> 16) & 1u);   // round-to-nearest-even
  return (unsigned short)(x >> 16);
}

// A fragment (16x32 bf16) from LDS row-major [rows][256].
// contiguous K order (same as prep_kernel B): lanes 0-15: row=lane, K = kbase+{0..15}; lanes 16-31: row=lane-16, K = kbase+{16..31}
__device__ __forceinline__ v16bf load_a_frag(const unsigned short* xsw, int lane, int kbase) {
  int r = lane & 15, half = lane >> 4;
  const unsigned short* p = xsw + r * 256 + kbase + half * 16;
  v8bf lo = *(const v8bf*)p;
  v8bf hi = *(const v8bf*)(p + 8);
  return __builtin_shufflevector(lo, hi, 0,1,2,3,4,5,6,7,8,9,10,11,12,13,14,15);
}

// B fragment: pre-swizzled contiguous 32B per lane
__device__ __forceinline__ v16bf load_b_frag(const unsigned short* wf, int kt, int nt, int NT, int lane) {
  const v16bf* w = (const v16bf*)wf;
  return w[(size_t)(kt * NT + nt) * 32 + lane];
}

// GEMM over two 16-row tiles sharing every B fragment (2x B reuse).
template<int KT, int NT>
__device__ __forceinline__ void gemm2(const unsigned short* xsw, const unsigned short* wf,
                                      int lane, v8f (&acc0)[NT], v8f (&acc1)[NT]) {
  #pragma unroll
  for (int nt = 0; nt < NT; ++nt) { acc0[nt] = {}; acc1[nt] = {}; }
  #pragma unroll
  for (int kt = 0; kt < KT; ++kt) {
    v16bf a0 = load_a_frag(xsw,            lane, kt * 32);
    v16bf a1 = load_a_frag(xsw + 16 * 256, lane, kt * 32);
    #pragma unroll
    for (int nt = 0; nt < NT; ++nt) {
      v16bf b = load_b_frag(wf, kt, nt, NT, lane);
      acc0[nt] = __builtin_amdgcn_wmma_f32_16x16x32_bf16(false, a0, false, b,
                                                         (short)0, acc0[nt], false, false);
      acc1[nt] = __builtin_amdgcn_wmma_f32_16x16x32_bf16(false, a1, false, b,
                                                         (short)0, acc1[nt], false, false);
    }
  }
}

// LayerNorm (+bias, gamma, beta) + ReLU entirely in registers.
// C tile layout: acc[t][j]: lanes 0-15 -> row j, col = t*16+lane; lanes 16-31 -> row j+8, col = t*16+lane-16.
// Row statistics reduce across the 16 lanes of each half via shfl_xor width 16.
template<int NT, typename F>
__device__ __forceinline__ void ln_relu(v8f (&acc)[NT], const float* bias, const float* g,
                                        const float* be, int lane, F emit) {
  const float C = (float)(NT * 16);
  int r = lane & 15;
  float s1[8], s2[8];
  #pragma unroll
  for (int j = 0; j < 8; ++j) { s1[j] = 0.f; s2[j] = 0.f; }
  #pragma unroll
  for (int t = 0; t < NT; ++t) {
    float bc = bias[t * 16 + r];
    #pragma unroll
    for (int j = 0; j < 8; ++j) {
      float v = acc[t][j] + bc;
      acc[t][j] = v;
      s1[j] += v; s2[j] += v * v;
    }
  }
  #pragma unroll
  for (int j = 0; j < 8; ++j) {
    #pragma unroll
    for (int m = 1; m < 16; m <<= 1) {
      s1[j] += __shfl_xor(s1[j], m, 16);
      s2[j] += __shfl_xor(s2[j], m, 16);
    }
  }
  float mean[8], scale[8];
  #pragma unroll
  for (int j = 0; j < 8; ++j) {
    mean[j] = s1[j] / C;
    float var = s2[j] / C - mean[j] * mean[j];
    scale[j] = rsqrtf(var + 1e-3f);
  }
  #pragma unroll
  for (int t = 0; t < NT; ++t) {
    float gc = g[t * 16 + r], bc2 = be[t * 16 + r];
    #pragma unroll
    for (int j = 0; j < 8; ++j) {
      float o = (acc[t][j] - mean[j]) * scale[j] * gc + bc2;
      o = fmaxf(o, 0.f);
      emit(t, j, o);
    }
  }
}

// ---------- prep: weights [K,N] f32 -> bf16 B-fragment layout ----------
// element idx = ((kt*NT+nt)*32 + lane)*16 + e ; k = kt*32 + (lane>=16?16:0) + e ; n = nt*16 + (lane&15)
__global__ void prep_kernel(const float* __restrict__ W, unsigned short* __restrict__ out,
                            int KT, int NT, int Ncols) {
  int total = KT * NT * 512;
  for (int idx = blockIdx.x * blockDim.x + threadIdx.x; idx < total; idx += gridDim.x * blockDim.x) {
    int e    = idx & 15;
    int lane = (idx >> 4) & 31;
    int ntk  = idx >> 9;
    int nt   = ntk % NT;
    int kt   = ntk / NT;
    int k    = kt * 32 + ((lane >> 4) << 4) + e;
    int n    = nt * 16 + (lane & 15);
    out[idx] = f2bf(W[(size_t)k * Ncols + n]);
  }
}

__global__ void zero_kernel(uint4* __restrict__ p, int n4) {
  uint4 z = {0u, 0u, 0u, 0u};
  for (int i = blockIdx.x * blockDim.x + threadIdx.x; i < n4; i += gridDim.x * blockDim.x)
    p[i] = z;
}

__global__ void iota_kernel(int* __restrict__ p, int n) {
  int i = blockIdx.x * blockDim.x + threadIdx.x;
  if (i < n) p[i] = i;
}

// ---------- point pipeline: x0 -> L1(128) -> L2(256) -> scatter-max + head(64) ----------
__global__ __launch_bounds__(WAVES * 32) void point_kernel(
    const float* __restrict__ features, const float* __restrict__ fcl,
    const int* __restrict__ coors,
    const float* __restrict__ b1, const float* __restrict__ g1, const float* __restrict__ be1,
    const float* __restrict__ b2, const float* __restrict__ g2, const float* __restrict__ be2,
    const float* __restrict__ ob, const float* __restrict__ og, const float* __restrict__ obe,
    const unsigned short* __restrict__ w1f, const unsigned short* __restrict__ w2f,
    const unsigned short* __restrict__ owf,
    float* __restrict__ outp, unsigned int* __restrict__ agg,
    int N, int ntiles32)
{
  __shared__ __align__(16) unsigned short xs[WAVES][32][256];
  __shared__ int cvox[WAVES][32];
  int wave = threadIdx.x >> 5;
  int lane = threadIdx.x & 31;
  int r = lane & 15, half = lane >> 4;
  unsigned short* xsw = &xs[wave][0][0];

  int tile = blockIdx.x * WAVES + wave;
  if (tile >= ntiles32) tile = ntiles32 - 1;
  int base = tile * 32;

  {
    int p = base + lane; if (p >= N) p = N - 1;
    cvox[wave][lane] = coors[p];
  }
  // build x0 = concat(features[125], f_cluster/XYZ_NORM[3]) as bf16 (coalesced: lane = column group)
  {
    int c0 = lane * 4;
    #pragma unroll 2
    for (int rr = 0; rr < 32; ++rr) {
      int p = base + rr; if (p >= N) p = N - 1;
      #pragma unroll
      for (int i = 0; i < 4; ++i) {
        int col = c0 + i;
        float v;
        if (col < 125) {
          v = features[(size_t)p * 125 + col];
        } else {
          int d = col - 125;
          float inv = (d == 2) ? 0.25f : 0.05f;
          v = fcl[(size_t)p * 3 + d] * inv;
        }
        xs[wave][rr][col] = f2bf(v);
      }
    }
  }
  __syncthreads();

  // Layer 1: [32,128] @ [128,128]
  {
    v8f acc0[8], acc1[8];
    gemm2<4, 8>(xsw, w1f, lane, acc0, acc1);
    __syncthreads();
    ln_relu<8>(acc0, b1, g1, be1, lane, [&](int t, int j, float o) {
      xs[wave][j + 8 * half][t * 16 + r] = f2bf(o);
    });
    ln_relu<8>(acc1, b1, g1, be1, lane, [&](int t, int j, float o) {
      xs[wave][16 + j + 8 * half][t * 16 + r] = f2bf(o);
    });
  }
  __syncthreads();

  // Layer 2: [32,128] @ [128,256]  + scatter-max into voxels
  {
    v8f acc0[16], acc1[16];
    gemm2<4, 16>(xsw, w2f, lane, acc0, acc1);
    __syncthreads();
    ln_relu<16>(acc0, b2, g2, be2, lane, [&](int t, int j, float o) {
      int row = j + 8 * half;
      int col = t * 16 + r;
      xs[wave][row][col] = f2bf(o);
      // post-ReLU values are >= 0: uint bit-pattern compare == float compare
      atomicMax(&agg[(size_t)cvox[wave][row] * 256 + col], __float_as_uint(o));
    });
    ln_relu<16>(acc1, b2, g2, be2, lane, [&](int t, int j, float o) {
      int row = 16 + j + 8 * half;
      int col = t * 16 + r;
      xs[wave][row][col] = f2bf(o);
      atomicMax(&agg[(size_t)cvox[wave][row] * 256 + col], __float_as_uint(o));
    });
  }
  __syncthreads();

  // Output head: [32,256] @ [256,64]
  {
    v8f acc0[4], acc1[4];
    gemm2<8, 4>(xsw, owf, lane, acc0, acc1);
    ln_relu<4>(acc0, ob, og, obe, lane, [&](int t, int j, float o) {
      int p = base + j + 8 * half; if (p >= N) p = N - 1;
      outp[(size_t)p * 64 + t * 16 + r] = o;
    });
    ln_relu<4>(acc1, ob, og, obe, lane, [&](int t, int j, float o) {
      int p = base + 16 + j + 8 * half; if (p >= N) p = N - 1;
      outp[(size_t)p * 64 + t * 16 + r] = o;
    });
  }
}

// ---------- voxel pipeline: agg -> P1(256) -> P2(256) ----------
__global__ __launch_bounds__(WAVES * 32) void voxel_kernel(
    const float* __restrict__ aggf,
    const float* __restrict__ pb1, const float* __restrict__ pg1, const float* __restrict__ pbe1,
    const float* __restrict__ pb2, const float* __restrict__ pg2, const float* __restrict__ pbe2,
    const unsigned short* __restrict__ pw1f, const unsigned short* __restrict__ pw2f,
    float* __restrict__ outv, int V, int ntiles32)
{
  __shared__ __align__(16) unsigned short xs[WAVES][32][256];
  int wave = threadIdx.x >> 5;
  int lane = threadIdx.x & 31;
  int r = lane & 15, half = lane >> 4;
  unsigned short* xsw = &xs[wave][0][0];

  int tile = blockIdx.x * WAVES + wave;
  if (tile >= ntiles32) tile = ntiles32 - 1;
  int base = tile * 32;

  // load agg tile (aligned float4, coalesced)
  {
    int c0 = lane * 4;
    #pragma unroll 2
    for (int rr = 0; rr < 32; ++rr) {
      int vx = base + rr; if (vx >= V) vx = V - 1;
      float4 v = *(const float4*)(aggf + (size_t)vx * 256 + c0);
      xs[wave][rr][c0 + 0] = f2bf(v.x);
      xs[wave][rr][c0 + 1] = f2bf(v.y);
      xs[wave][rr][c0 + 2] = f2bf(v.z);
      xs[wave][rr][c0 + 3] = f2bf(v.w);
    }
  }
  __syncthreads();

  {
    v8f acc0[16], acc1[16];
    gemm2<8, 16>(xsw, pw1f, lane, acc0, acc1);
    __syncthreads();
    ln_relu<16>(acc0, pb1, pg1, pbe1, lane, [&](int t, int j, float o) {
      xs[wave][j + 8 * half][t * 16 + r] = f2bf(o);
    });
    ln_relu<16>(acc1, pb1, pg1, pbe1, lane, [&](int t, int j, float o) {
      xs[wave][16 + j + 8 * half][t * 16 + r] = f2bf(o);
    });
  }
  __syncthreads();

  {
    v8f acc0[16], acc1[16];
    gemm2<8, 16>(xsw, pw2f, lane, acc0, acc1);
    ln_relu<16>(acc0, pb2, pg2, pbe2, lane, [&](int t, int j, float o) {
      int vx = base + j + 8 * half; if (vx >= V) vx = V - 1;
      outv[(size_t)vx * 256 + t * 16 + r] = o;
    });
    ln_relu<16>(acc1, pb2, pg2, pbe2, lane, [&](int t, int j, float o) {
      int vx = base + 16 + j + 8 * half; if (vx >= V) vx = V - 1;
      outv[(size_t)vx * 256 + t * 16 + r] = o;
    });
  }
}

// ---------- host launcher ----------
extern "C" void kernel_launch(void* const* d_in, const int* in_sizes, int n_in,
                              void* d_out, int out_size, void* d_ws, size_t ws_size,
                              hipStream_t stream) {
  // setup_inputs order:
  // 0 points, 1 features, 2 coors, 3 f_cluster,
  // 4 w1, 5 b1, 6 g1, 7 be1, 8 w2, 9 b2, 10 g2, 11 be2,
  // 12 pw1, 13 pb1, 14 pg1, 15 pbe1, 16 pw2, 17 pb2, 18 pg2, 19 pbe2,
  // 20 ow, 21 ob, 22 og, 23 obe
  const float* features = (const float*)d_in[1];
  const int*   coors    = (const int*)d_in[2];
  const float* fcl      = (const float*)d_in[3];
  const float* w1  = (const float*)d_in[4];
  const float* b1  = (const float*)d_in[5];
  const float* g1  = (const float*)d_in[6];
  const float* be1 = (const float*)d_in[7];
  const float* w2  = (const float*)d_in[8];
  const float* b2  = (const float*)d_in[9];
  const float* g2  = (const float*)d_in[10];
  const float* be2 = (const float*)d_in[11];
  const float* pw1 = (const float*)d_in[12];
  const float* pb1 = (const float*)d_in[13];
  const float* pg1 = (const float*)d_in[14];
  const float* pbe1= (const float*)d_in[15];
  const float* pw2 = (const float*)d_in[16];
  const float* pb2 = (const float*)d_in[17];
  const float* pg2 = (const float*)d_in[18];
  const float* pbe2= (const float*)d_in[19];
  const float* ow  = (const float*)d_in[20];
  const float* ob  = (const float*)d_in[21];
  const float* og  = (const float*)d_in[22];
  const float* obe = (const float*)d_in[23];

  const int N = in_sizes[2];      // 500000
  const int V = 60000;

  char* ws = (char*)d_ws;
  unsigned short* w1f  = (unsigned short*)ws;      // 128*128
  unsigned short* w2f  = w1f  + 128 * 128;         // 128*256
  unsigned short* owf  = w2f  + 128 * 256;         // 256*64
  unsigned short* pw1f = owf  + 256 * 64;          // 256*256
  unsigned short* pw2f = pw1f + 256 * 256;         // 256*256
  unsigned int*  agg   = (unsigned int*)(ws + (4u << 20));   // V*256 f32 (bit-pattern max)

  // weight swizzle (bf16 B-fragment layout)
  prep_kernel<<<64,  256, 0, stream>>>(w1,  w1f,  4, 8,  128);
  prep_kernel<<<128, 256, 0, stream>>>(w2,  w2f,  4, 16, 256);
  prep_kernel<<<64,  256, 0, stream>>>(ow,  owf,  8, 4,  64);
  prep_kernel<<<256, 256, 0, stream>>>(pw1, pw1f, 8, 16, 256);
  prep_kernel<<<256, 256, 0, stream>>>(pw2, pw2f, 8, 16, 256);

  // zero agg every call (graph-replay safe; also implements where(occ,...,0))
  zero_kernel<<<2048, 256, 0, stream>>>((uint4*)agg, V * 256 / 4);

  int ntp = (N + 31) / 32;
  point_kernel<<<(ntp + WAVES - 1) / WAVES, WAVES * 32, 0, stream>>>(
      features, fcl, coors,
      b1, g1, be1, b2, g2, be2, ob, og, obe,
      w1f, w2f, owf,
      (float*)d_out, agg, N, ntp);

  int ntv = (V + 31) / 32;
  voxel_kernel<<<(ntv + WAVES - 1) / WAVES, WAVES * 32, 0, stream>>>(
      (const float*)agg,
      pb1, pg1, pbe1, pb2, pg2, pbe2,
      pw1f, pw2f,
      (float*)d_out + (size_t)N * 64, V, ntv);

  iota_kernel<<<(V + 255) / 256, 256, 0, stream>>>(
      (int*)((float*)d_out + (size_t)N * 64 + (size_t)V * 256), V);
}
